// RNASampler_48473000903424
// MI455X (gfx1250) — matrix operand order repaired and checked by reading, not compiled, not run on hardware
//
#include <hip/hip_runtime.h>
#include <hip/hip_bf16.h>
#include <stdint.h>

typedef __attribute__((ext_vector_type(16))) _Float16 v16h;
typedef __attribute__((ext_vector_type(8)))  _Float16 v8h;
typedef __attribute__((ext_vector_type(8)))  float    v8f;

#define L_RES   2048
#define NSTEPS  10
#define SEG     5
#define MIN_D   3.0f
#define EPSF    1e-8f
#define PSTRIDE 32   // floats per step in params table

// ---------------- deterministic counter RNG (splitmix64 + Box-Muller) -------
__device__ __forceinline__ unsigned long long sm64(unsigned long long x) {
  x += 0x9E3779B97F4A7C15ull;
  x = (x ^ (x >> 30)) * 0xBF58476D1CE4E5B9ull;
  x = (x ^ (x >> 27)) * 0x94D049BB133111EBull;
  return x ^ (x >> 31);
}
__device__ __forceinline__ float rng_u01(unsigned long long ctr) {
  return (float)(sm64(ctr) >> 40) * (1.0f / 16777216.0f);   // [0,1)
}
__device__ __forceinline__ float rng_normal(unsigned long long ctr) {
  float u1 = fmaxf(rng_u01(ctr * 2u + 1u), 1e-7f);
  float u2 = rng_u01(ctr * 2u + 2u);
  return sqrtf(-2.0f * logf(u1)) * cosf(6.28318530718f * u2);
}

// ---------------- conversion / transpose helpers ----------------------------
__global__ void k_cvt_f16(const float* __restrict__ in, _Float16* __restrict__ out, int n) {
  int i = blockIdx.x * blockDim.x + threadIdx.x;
  if (i < n) out[i] = (_Float16)in[i];
}

// W: (K x N) row-major f32  ->  Wt: (Npad x N... actually Npad x K) f16, zero-padded cols
__global__ void k_transpose_f16(const float* __restrict__ W, _Float16* __restrict__ Wt,
                                int K, int N, int Npad) {
  int i = blockIdx.x * blockDim.x + threadIdx.x;
  if (i >= Npad * K) return;
  int np = i / K, k = i % K;
  float v = (np < N) ? W[(size_t)k * N + np] : 0.0f;
  Wt[(size_t)np * K + k] = (_Float16)v;
}

// ---------------- WMMA GEMM: Out = act(A(MxK,f16) * Bt(N x K,f16)^T + bias) -
// One wave computes one 16x16 tile via v_wmma_f32_16x16x32_f16.
// A layout per ISA: lane = row(l&15); k-half = (l>>4)*8 ; halves 8..15 at +16.
// Bt pre-transposed, same K layout as A: col=(l&15), k-half=(l>>4)*8; halves 8..15 at +16.
__global__ void k_gemm_wmma(const _Float16* __restrict__ A,
                            const _Float16* __restrict__ Bt,
                            const float*    __restrict__ bias,
                            void*           __restrict__ Out,
                            int M, int N, int K, int Nout, int relu, int outF16) {
  int wave = (blockIdx.x * blockDim.x + threadIdx.x) >> 5;
  int lane = threadIdx.x & 31;
  int ntilesN = N >> 4;
  int tiles   = (M >> 4) * ntilesN;
  if (wave >= tiles) return;
  int mt = wave / ntilesN;
  int nt = wave % ntilesN;

  int half = lane >> 4;                 // 0 or 1
  int row  = (mt << 4) + (lane & 15);   // A row for this lane
  int n    = (nt << 4) + (lane & 15);   // output column for this lane

  v8f c;
  float bv = (n < Nout) ? bias[n] : 0.0f;
#pragma unroll
  for (int v = 0; v < 8; ++v) c[v] = bv;

  const _Float16* arow = A  + (size_t)row * K + half * 8;
  const _Float16* bcol = Bt + (size_t)n   * K + half * 8;

  for (int k0 = 0; k0 < K; k0 += 32) {
    v8h alo = *(const v8h*)(arow + k0);        // K = k0+kb .. +7
    v8h ahi = *(const v8h*)(arow + k0 + 16);   // K = k0+kb+16 .. +23
    v8h blo = *(const v8h*)(bcol + k0);        // K = k0+kb .. +7
    v8h bhi = *(const v8h*)(bcol + k0 + 16);   // K = k0+kb+16 .. +23
    v16h a = __builtin_shufflevector(alo, ahi, 0,1,2,3,4,5,6,7,8,9,10,11,12,13,14,15);
    v16h b = __builtin_shufflevector(blo, bhi, 0,1,2,3,4,5,6,7,8,9,10,11,12,13,14,15);
    c = __builtin_amdgcn_wmma_f32_16x16x32_f16(false, a, false, b, (short)0, c, false, false);
  }

  int mbase = (mt << 4) + (half << 3);  // C layout: VGPR v -> M = v + 8*half
  if (outF16) {
    _Float16* O = (_Float16*)Out;
#pragma unroll
    for (int v = 0; v < 8; ++v) {
      float x = c[v];
      if (relu && x < 0.0f) x = 0.0f;
      O[(size_t)(mbase + v) * Nout + n] = (_Float16)x;
    }
  } else if (n < Nout) {
    float* O = (float*)Out;
#pragma unroll
    for (int v = 0; v < 8; ++v) {
      float x = c[v];
      if (relu && x < 0.0f) x = 0.0f;
      O[(size_t)(mbase + v) * Nout + n] = x;
    }
  }
}

// ---------------- per-step random parameters + rotation matrices ------------
__global__ void k_params(float* __restrict__ params) {
  if (blockIdx.x != 0 || threadIdx.x != 0) return;
  for (int s = 0; s < NSTEPS; ++s) {
    float* p = params + s * PSTRIDE;
    unsigned long long base = 0x5EEDBA5Eull + (unsigned long long)s * 131ull;
    int maxstart = L_RES - 10;                                   // inclusive
    int st = (int)(rng_u01(base + 1) * (float)(maxstart + 1));
    if (st > maxstart) st = maxstart;
    p[0] = (float)st;
    p[1] = (rng_u01(base + 2) < 0.3f) ? 1.0f : 0.0f;             // do_motif
    int mi = (int)(rng_u01(base + 3) * 4.0f); if (mi > 3) mi = 3;
    p[2] = (float)mi;
    p[3] = (rng_u01(base + 4) < 0.5f) ? 1.0f : 0.0f;             // do_rot
    float q[4]; float nq = 0.0f;
    for (int i = 0; i < 4; ++i) { q[i] = rng_normal(base + 5 + i); nq += q[i] * q[i]; }
    nq = sqrtf(nq) + EPSF;
    for (int i = 0; i < 4; ++i) { q[i] /= nq; p[4 + i] = q[i]; }
    p[8] = (rng_u01(base + 9) < 0.5f) ? 1.0f : 0.0f;             // do_trans
    for (int i = 0; i < 3; ++i) p[9 + i] = rng_normal(base + 10 + i) * 0.5f;
    float w = q[0], x = q[1], y = q[2], z = q[3];
    float* R = p + 12;                                           // row-major R[d][c]
    R[0] = 1 - 2*(y*y + z*z); R[1] = 2*(x*y - w*z); R[2] = 2*(x*z + w*y);
    R[3] = 2*(x*y + w*z);     R[4] = 1 - 2*(x*x + z*z); R[5] = 2*(y*z - w*x);
    R[6] = 2*(x*z - w*y);     R[7] = 2*(y*z + w*x);     R[8] = 1 - 2*(x*x + y*y);
  }
}

// ---------------- coords init: coords = mlp_out + 0.1*noise -----------------
__global__ void k_init_coords(const float* __restrict__ c0, float* __restrict__ coords, int n) {
  int i = blockIdx.x * blockDim.x + threadIdx.x;
  if (i < n) coords[i] = c0[i] + 0.1f * rng_normal(0xABCDEF00ull + (unsigned long long)i);
}

// ---------------- motif / rotate / translate (in-place, per-atom thread) ----
__global__ void k_mrt(float* __restrict__ coords, const float* __restrict__ params,
                      const float* __restrict__ motifs, int step) {
  int idx = blockIdx.x * blockDim.x + threadIdx.x;
  if (idx >= L_RES * 3) return;
  int l = idx / 3, a = idx % 3;
  const float* p = params + step * PSTRIDE;
  float x = coords[idx * 3 + 0], y = coords[idx * 3 + 1], z = coords[idx * 3 + 2];
  int st = (int)p[0];
  if (p[1] > 0.5f && l >= st && l < st + SEG) {
    int mi = (int)p[2];
    x = motifs[mi * 9 + a * 3 + 0];
    y = motifs[mi * 9 + a * 3 + 1];
    z = motifs[mi * 9 + a * 3 + 2];
  }
  if (p[3] > 0.5f) {                   // rot[d] = sum_c v[c] * R[d][c]
    const float* R = p + 12;
    float nx = R[0]*x + R[1]*y + R[2]*z;
    float ny = R[3]*x + R[4]*y + R[5]*z;
    float nz = R[6]*x + R[7]*y + R[8]*z;
    x = nx; y = ny; z = nz;
  }
  if (p[8] > 0.5f) { x += p[9]; y += p[10]; z += p[11]; }
  coords[idx * 3 + 0] = x; coords[idx * 3 + 1] = y; coords[idx * 3 + 2] = z;
}

// ---------------- bond constraint: read A, write B --------------------------
__global__ void k_bond(const float* __restrict__ in, float* __restrict__ out) {
  int idx = blockIdx.x * blockDim.x + threadIdx.x;
  if (idx >= L_RES * 3) return;
  int l = idx / 3, a = idx % 3;
  float x = in[idx * 3 + 0], y = in[idx * 3 + 1], z = in[idx * 3 + 2];
  if (l < L_RES - 1) {
    int j = ((l + 1) * 3 + a) * 3;
    float dx = in[j + 0] - x, dy = in[j + 1] - y, dz = in[j + 2] - z;
    float d = sqrtf(dx * dx + dy * dy + dz * dz);
    if (d < MIN_D) {
      float f = (MIN_D - d) / (d + EPSF);
      x += dx * f; y += dy * f; z += dz * f;
    }
  }
  out[idx * 3 + 0] = x; out[idx * 3 + 1] = y; out[idx * 3 + 2] = z;
}

// ---------------- repulsion: one block per column j -------------------------
__global__ void k_rep(const float* __restrict__ coords, float* __restrict__ rep_new) {
  __shared__ int sbest[256];
  int j = blockIdx.x;
  float jx = coords[j * 9 + 0], jy = coords[j * 9 + 1], jz = coords[j * 9 + 2];
  int best = -1;
  for (int i = threadIdx.x; i < j; i += blockDim.x) {
    float dx = jx - coords[i * 9 + 0];
    float dy = jy - coords[i * 9 + 1];
    float dz = jz - coords[i * 9 + 2];
    float d = sqrtf(dx * dx + dy * dy + dz * dz);
    if (d < MIN_D) best = i;          // i increases, keeps max violating i
  }
  sbest[threadIdx.x] = best;
  __syncthreads();
  for (int s = blockDim.x >> 1; s > 0; s >>= 1) {
    if (threadIdx.x < s) sbest[threadIdx.x] = max(sbest[threadIdx.x], sbest[threadIdx.x + s]);
    __syncthreads();
  }
  if (threadIdx.x == 0) {
    int b = sbest[0];
    float nx = jx, ny = jy, nz = jz;
    if (b >= 0) {
      float bx = coords[b * 9 + 0], by = coords[b * 9 + 1], bz = coords[b * 9 + 2];
      float dx = jx - bx, dy = jy - by, dz = jz - bz;
      float d = sqrtf(dx * dx + dy * dy + dz * dz);
      float f = (MIN_D - d) / (d + EPSF);
      nx = bx + dx * (1.0f + f) - dx;  // = bx + dx*f ... keep explicit below
      nx = bx + dx * f; ny = by + dy * f; nz = bz + dz * f;
    }
    rep_new[j * 3 + 0] = nx; rep_new[j * 3 + 1] = ny; rep_new[j * 3 + 2] = nz;
  }
}

// ---------------- merge rep row back: B + rep -> A --------------------------
__global__ void k_apply(const float* __restrict__ bondbuf, const float* __restrict__ rep_new,
                        float* __restrict__ outc) {
  int idx = blockIdx.x * blockDim.x + threadIdx.x;
  if (idx >= L_RES * 9) return;
  int l = idx / 9, r = idx % 9;
  float v = bondbuf[idx];
  if (r < 3) v = rep_new[l * 3 + r];
  outc[idx] = v;
}

__global__ void k_copy(const float* __restrict__ in, float* __restrict__ out, int n) {
  int i = blockIdx.x * blockDim.x + threadIdx.x;
  if (i < n) out[i] = in[i];
}

// ---------------------------------------------------------------------------
extern "C" void kernel_launch(void* const* d_in, const int* in_sizes, int n_in,
                              void* d_out, int out_size, void* d_ws, size_t ws_size,
                              hipStream_t stream) {
  (void)in_sizes; (void)n_in; (void)out_size; (void)ws_size;
  const float* E    = (const float*)d_in[0];  // (1,2048,512)
  const float* W1   = (const float*)d_in[1];  // (512,256)
  const float* b1   = (const float*)d_in[2];
  const float* W2   = (const float*)d_in[3];  // (256,128)
  const float* b2   = (const float*)d_in[4];
  const float* W3   = (const float*)d_in[5];  // (128,9)
  const float* b3   = (const float*)d_in[6];
  const float* mot  = (const float*)d_in[7];  // (4,3,3)
  float* outp = (float*)d_out;

  const int M = L_RES, D = 512, H1N = 256, H2N = 128;

  // workspace carve-up (256B aligned)
  char* base = (char*)d_ws;
  size_t off = 0;
  auto carve = [&](size_t bytes) -> char* {
    char* p = base + off;
    off = (off + bytes + 255) & ~(size_t)255;
    return p;
  };
  _Float16* E16  = (_Float16*)carve((size_t)M * D   * 2);
  _Float16* W1t  = (_Float16*)carve((size_t)H1N * D * 2);
  _Float16* H1   = (_Float16*)carve((size_t)M * H1N * 2);
  _Float16* W2t  = (_Float16*)carve((size_t)H2N * H1N * 2);
  _Float16* H2   = (_Float16*)carve((size_t)M * H2N * 2);
  _Float16* W3t  = (_Float16*)carve((size_t)16 * H2N * 2);
  float*    C0   = (float*)carve((size_t)M * 9 * 4);
  float*    cA   = (float*)carve((size_t)M * 9 * 4);
  float*    cB   = (float*)carve((size_t)M * 9 * 4);
  float*    repN = (float*)carve((size_t)M * 3 * 4);
  float*    prm  = (float*)carve((size_t)NSTEPS * PSTRIDE * 4);

  // 1) convert inputs to f16 (+ weight transposes, zero-padded for GEMM3)
  {
    int n = M * D;
    k_cvt_f16<<<(n + 255) / 256, 256, 0, stream>>>(E, E16, n);
  }
  k_transpose_f16<<<(H1N * D   + 255) / 256, 256, 0, stream>>>(W1, W1t, D,   H1N, H1N);
  k_transpose_f16<<<(H2N * H1N + 255) / 256, 256, 0, stream>>>(W2, W2t, H1N, H2N, H2N);
  k_transpose_f16<<<(16  * H2N + 255) / 256, 256, 0, stream>>>(W3, W3t, H2N, 9,   16);

  // 2) MLP via WMMA (one wave / 16x16 tile; 128 threads = 4 waves per block)
  {
    int tiles = (M / 16) * (H1N / 16);
    k_gemm_wmma<<<(tiles * 32 + 127) / 128, 128, 0, stream>>>(
        E16, W1t, b1, (void*)H1, M, H1N, D, H1N, /*relu=*/1, /*f16=*/1);
  }
  {
    int tiles = (M / 16) * (H2N / 16);
    k_gemm_wmma<<<(tiles * 32 + 127) / 128, 128, 0, stream>>>(
        H1, W2t, b2, (void*)H2, M, H2N, H1N, H2N, /*relu=*/1, /*f16=*/1);
  }
  {
    int tiles = (M / 16) * 1;
    k_gemm_wmma<<<(tiles * 32 + 127) / 128, 128, 0, stream>>>(
        H2, W3t, b3, (void*)C0, M, 16, H2N, 9, /*relu=*/0, /*f16=*/0);
  }

  // 3) random step parameters + initial noise
  k_params<<<1, 1, 0, stream>>>(prm);
  {
    int n = M * 9;
    k_init_coords<<<(n + 255) / 256, 256, 0, stream>>>(C0, cA, n);
  }

  // 4) 10 sequential edit+constraint steps
  for (int s = 0; s < NSTEPS; ++s) {
    k_mrt  <<<(M * 3 + 255) / 256, 256, 0, stream>>>(cA, prm, mot, s);
    k_bond <<<(M * 3 + 255) / 256, 256, 0, stream>>>(cA, cB);
    k_rep  <<<M, 256, 0, stream>>>(cB, repN);
    k_apply<<<(M * 9 + 255) / 256, 256, 0, stream>>>(cB, repN, cA);
  }

  // 5) output
  {
    int n = M * 9;
    k_copy<<<(n + 255) / 256, 256, 0, stream>>>(cA, outp, n);
  }
}
